// MambaBlock3D_26147760898623
// MI455X (gfx1250) — matrix operand order repaired and checked by reading, not compiled, not run on hardware
//
#include <hip/hip_runtime.h>

// ---------------- problem constants (from reference) ----------------
#define L_SEQ   8192          // 8*32*32 flattened sequence
#define C_DIM   128           // d_model
#define DIN     256           // d_inner
#define NST     16            // d_state
#define DTRANK  8
#define NPROJ   40            // DTRANK + 2*NST
#define NPROJP  48            // padded to 3 WMMA n-tiles
#define NCHUNK  16
#define CHUNK   512           // NCHUNK*CHUNK == L_SEQ

typedef __attribute__((ext_vector_type(16))) _Float16 v16h;
typedef __attribute__((ext_vector_type(8)))  _Float16 v8h;
typedef __attribute__((ext_vector_type(8)))  float    v8f;

static __device__ __forceinline__ float silu_f(float v) {
    return v / (1.0f + __expf(-v));
}

// ---------------- small conversion / precompute kernels ----------------
__global__ void mamba_cvt_f16(const float* __restrict__ src, _Float16* __restrict__ dst, int n) {
    int i = blockIdx.x * blockDim.x + threadIdx.x;
    if (i < n) dst[i] = (_Float16)src[i];
}

// x_proj weights [40,256] -> padded f16 [48,256] (rows 40..47 zero)
__global__ void mamba_cvt_xproj(const float* __restrict__ src, _Float16* __restrict__ dst) {
    int i = blockIdx.x * blockDim.x + threadIdx.x;
    if (i < NPROJP * DIN) {
        int j = i / DIN, k = i % DIN;
        dst[i] = (j < NPROJ) ? (_Float16)src[j * DIN + k] : (_Float16)0.0f;
    }
}

// A = -exp(A_log), [256,16]
__global__ void mamba_aneg(const float* __restrict__ alog, float* __restrict__ aneg) {
    int i = blockIdx.x * blockDim.x + threadIdx.x;
    if (i < DIN * NST) aneg[i] = -__expf(alog[i]);
}

// ---------------- layernorm: x[c,l] -> xs_f16[l,c] ----------------
__global__ void mamba_layernorm(const float* __restrict__ x,
                                const float* __restrict__ g,
                                const float* __restrict__ b,
                                _Float16* __restrict__ xs) {
    int l = blockIdx.x * blockDim.x + threadIdx.x;     // coalesced over l
    if (l >= L_SEQ) return;
    float sum = 0.f, sq = 0.f;
    for (int c = 0; c < C_DIM; ++c) {
        float v = x[(size_t)c * L_SEQ + l];
        sum += v; sq += v * v;
    }
    float mu = sum * (1.0f / C_DIM);
    float var = sq * (1.0f / C_DIM) - mu * mu;
    float rs = rsqrtf(var + 1e-5f);
    for (int c = 0; c < C_DIM; ++c) {
        float v = (x[(size_t)c * L_SEQ + l] - mu) * rs * g[c] + b[c];
        xs[(size_t)l * C_DIM + c] = (_Float16)v;
    }
}

// ---------------- WMMA GEMM: C[M,N] = A[M,K](f16) * B[N,K](f16)^T ----------------
// Each wave: one 16-row M tile x NT 16-col N tiles (A fragment reused NT times,
// NT independent WMMAs per K step keep the XDL pipe fed). 4 waves/block over M.
// Fragment layouts used here (contiguous k, same order in A and B):
//   A 16x32 f16 : lane m=l&15, hi=l>>4; elem  j   -> k=hi*16+j  (contiguous in A[M,K])
//   B 32x16 f16 : lane n=l&15, hi=l>>4; elem  j   -> k=hi*16+j  (contiguous in W[N,K])
//   C/D f32     : VGPR r: lanes0-15 M=r, lanes16-31 M=8+r; N=lane&15
template <int K, int NT, bool TSTORE>
__global__ __launch_bounds__(128) void mamba_gemm_wmma(const _Float16* __restrict__ A,
                                                       const _Float16* __restrict__ B,
                                                       float* __restrict__ C,
                                                       int M, int N, int ldc) {
    int wave = threadIdx.x >> 5;
    int lane = threadIdx.x & 31;
    int row0 = (blockIdx.x * 4 + wave) * 16;
    int col0 = blockIdx.y * (16 * NT);
    if (row0 >= M || col0 >= N) return;

    int m  = lane & 15;
    int hi = lane >> 4;
    const _Float16* arow = A + (size_t)(row0 + m) * K;
    const _Float16* brow[NT];
#pragma unroll
    for (int t = 0; t < NT; ++t)
        brow[t] = B + (size_t)(col0 + t * 16 + m) * K + hi * 16;

    v8f acc[NT];
#pragma unroll
    for (int t = 0; t < NT; ++t) acc[t] = (v8f){};

#pragma unroll
    for (int k0 = 0; k0 < K; k0 += 32) {
        v16h a = *(const v16h*)(arow + k0 + hi * 16);
#pragma unroll
        for (int t = 0; t < NT; ++t) {
            v16h bf = *(const v16h*)(brow[t] + k0);
            acc[t] = __builtin_amdgcn_wmma_f32_16x16x32_f16(false, a, false, bf,
                                                            (short)0, acc[t], false, false);
        }
    }
#pragma unroll
    for (int t = 0; t < NT; ++t) {
#pragma unroll
        for (int r = 0; r < 8; ++r) {
            int row = row0 + hi * 8 + r;
            int col = col0 + t * 16 + m;
            if (TSTORE)  C[(size_t)col * M + row] = acc[t][r];   // output [C, L]
            else         C[(size_t)row * ldc + col] = acc[t][r];
        }
    }
}

// ---------------- causal depthwise conv(4) + bias + SiLU ----------------
// xz[l,512]: cols 0..255 = xin_raw, cols 256..511 = z (left untouched here)
__global__ void mamba_conv_silu(const float* __restrict__ xz,
                                const float* __restrict__ cw,
                                const float* __restrict__ cb,
                                float* __restrict__ xo_f32,
                                _Float16* __restrict__ xo_f16) {
    int l = blockIdx.x;
    int d = threadIdx.x;
    float acc = cb[d];
    float w0 = cw[d * 4 + 0], w1 = cw[d * 4 + 1], w2 = cw[d * 4 + 2], w3 = cw[d * 4 + 3];
    if (l >= 3) acc += w0 * xz[(size_t)(l - 3) * 512 + d];
    if (l >= 2) acc += w1 * xz[(size_t)(l - 2) * 512 + d];
    if (l >= 1) acc += w2 * xz[(size_t)(l - 1) * 512 + d];
    acc += w3 * xz[(size_t)l * 512 + d];
    float s = silu_f(acc);
    xo_f32[(size_t)l * DIN + d] = s;
    xo_f16[(size_t)l * DIN + d] = (_Float16)s;
}

// ---------------- dt_proj (K=8, scalar) + softplus ----------------
__global__ void mamba_dtproj(const float* __restrict__ dbl,
                             const float* __restrict__ wdt,
                             const float* __restrict__ bdt,
                             float* __restrict__ dtf) {
    int l = blockIdx.x;
    int d = threadIdx.x;
    float acc = bdt[d];
#pragma unroll
    for (int j = 0; j < DTRANK; ++j)
        acc += dbl[(size_t)l * NPROJP + j] * wdt[d * DTRANK + j];
    float sp = (acc > 20.0f) ? acc : log1pf(__expf(acc));
    dtf[(size_t)l * DIN + d] = sp;
}

// ---------------- scan pass 1: per-chunk carries (aprod, bacc) ----------------
__global__ void mamba_scan_carry(const float* __restrict__ dtf,
                                 const float* __restrict__ xin,
                                 const float* __restrict__ dbl,
                                 const float* __restrict__ aneg,
                                 float* __restrict__ cA,
                                 float* __restrict__ cB) {
    int d = blockIdx.x;
    int ch = blockIdx.y;
    int lane = threadIdx.x & 31;
    int s = lane & 15;                          // lanes 16..31 duplicate lanes 0..15
    float Ad = aneg[d * NST + s];
    float ap = 1.0f, bc = 0.0f;
    int l0 = ch * CHUNK;
    for (int l = l0; l < l0 + CHUNK; ++l) {
        if (l + 16 < L_SEQ) {                   // cover latency of the serial chain
            __builtin_prefetch(&dtf[(size_t)(l + 16) * DIN + d], 0, 1);
            __builtin_prefetch(&dbl[(size_t)(l + 16) * NPROJP + DTRANK + s], 0, 1);
        }
        float dt = dtf[(size_t)l * DIN + d];
        float xv = xin[(size_t)l * DIN + d];
        float Bv = dbl[(size_t)l * NPROJP + DTRANK + s];
        float dA = __expf(dt * Ad);
        bc = dA * bc + dt * Bv * xv;
        ap *= dA;
    }
    if (lane < 16) {
        size_t idx = ((size_t)d * NCHUNK + ch) * NST + s;
        cA[idx] = ap;
        cB[idx] = bc;
    }
}

// ---------------- scan pass 2: exclusive prefix over chunks (4096 chains) ----------------
__global__ void mamba_scan_prefix(const float* __restrict__ cA,
                                  const float* __restrict__ cB,
                                  float* __restrict__ h0) {
    int t = blockIdx.x * blockDim.x + threadIdx.x;
    if (t >= DIN * NST) return;
    int d = t >> 4, s = t & 15;
    float h = 0.0f;
    for (int ch = 0; ch < NCHUNK; ++ch) {
        size_t idx = ((size_t)d * NCHUNK + ch) * NST + s;
        h0[idx] = h;
        h = cA[idx] * h + cB[idx];
    }
}

// ---------------- scan pass 3: y = sum_s h*C + D*xin, gate with silu(z) ----------------
__global__ void mamba_scan_final(const float* __restrict__ dtf,
                                 const float* __restrict__ xin,
                                 const float* __restrict__ dbl,
                                 const float* __restrict__ aneg,
                                 const float* __restrict__ h0,
                                 const float* __restrict__ xz,
                                 const float* __restrict__ Dp,
                                 _Float16* __restrict__ yg) {
    int d = blockIdx.x;
    int ch = blockIdx.y;
    int lane = threadIdx.x & 31;
    int s = lane & 15;
    float Ad = aneg[d * NST + s];
    float Dd = Dp[d];
    float h = h0[((size_t)d * NCHUNK + ch) * NST + s];
    int l0 = ch * CHUNK;
    for (int l = l0; l < l0 + CHUNK; ++l) {
        if (l + 16 < L_SEQ) {
            __builtin_prefetch(&dtf[(size_t)(l + 16) * DIN + d], 0, 1);
            __builtin_prefetch(&dbl[(size_t)(l + 16) * NPROJP + DTRANK + s], 0, 1);
        }
        float dt = dtf[(size_t)l * DIN + d];
        float xv = xin[(size_t)l * DIN + d];
        float Bv = dbl[(size_t)l * NPROJP + DTRANK + s];
        float Cv = dbl[(size_t)l * NPROJP + DTRANK + NST + s];
        float dA = __expf(dt * Ad);
        h = dA * h + dt * Bv * xv;
        float y = h * Cv;
        // reduce over 16 states (xor masks stay within each 16-lane half)
        y += __shfl_xor(y, 1);
        y += __shfl_xor(y, 2);
        y += __shfl_xor(y, 4);
        y += __shfl_xor(y, 8);
        if (lane == 0) {
            float zv = xz[(size_t)l * 512 + 256 + d];
            float out = (y + Dd * xv) * silu_f(zv);
            yg[(size_t)l * DIN + d] = (_Float16)out;
        }
    }
}

// ---------------- host launcher ----------------
extern "C" void kernel_launch(void* const* d_in, const int* in_sizes, int n_in,
                              void* d_out, int out_size, void* d_ws, size_t ws_size,
                              hipStream_t stream) {
    const float* x         = (const float*)d_in[0];
    const float* ln_w      = (const float*)d_in[1];
    const float* ln_b      = (const float*)d_in[2];
    const float* in_proj_w = (const float*)d_in[3];
    const float* conv_w    = (const float*)d_in[4];
    const float* conv_b    = (const float*)d_in[5];
    const float* x_proj_w  = (const float*)d_in[6];
    const float* dt_proj_w = (const float*)d_in[7];
    const float* dt_proj_b = (const float*)d_in[8];
    const float* A_log     = (const float*)d_in[9];
    const float* D_param   = (const float*)d_in[10];
    const float* out_proj_w= (const float*)d_in[11];
    float* out = (float*)d_out;

    char* wp = (char*)d_ws;
    auto alloc = [&](size_t bytes) -> void* {
        void* p = (void*)wp;
        wp += (bytes + 255) & ~(size_t)255;
        return p;
    };
    _Float16* w_in_h  = (_Float16*)alloc((size_t)512 * 128 * 2);
    _Float16* w_x_h   = (_Float16*)alloc((size_t)NPROJP * DIN * 2);
    _Float16* w_out_h = (_Float16*)alloc((size_t)C_DIM * DIN * 2);
    _Float16* xs_h    = (_Float16*)alloc((size_t)L_SEQ * C_DIM * 2);
    float*    xz      = (float*)   alloc((size_t)L_SEQ * 512 * 4);
    float*    xin_f   = (float*)   alloc((size_t)L_SEQ * DIN * 4);
    _Float16* xin_h   = (_Float16*)alloc((size_t)L_SEQ * DIN * 2);
    float*    dbl     = (float*)   alloc((size_t)L_SEQ * NPROJP * 4);
    float*    dtf     = (float*)   alloc((size_t)L_SEQ * DIN * 4);
    float*    aneg    = (float*)   alloc((size_t)DIN * NST * 4);
    float*    cA      = (float*)   alloc((size_t)DIN * NCHUNK * NST * 4);
    float*    cB      = (float*)   alloc((size_t)DIN * NCHUNK * NST * 4);
    float*    h0      = (float*)   alloc((size_t)DIN * NCHUNK * NST * 4);
    _Float16* yg      = (_Float16*)alloc((size_t)L_SEQ * DIN * 2);
    (void)ws_size; (void)n_in; (void)in_sizes; (void)out_size;

    // weight conversions + A precompute
    mamba_cvt_f16  <<<(512 * 128 + 255) / 256, 256, 0, stream>>>(in_proj_w, w_in_h, 512 * 128);
    mamba_cvt_xproj<<<(NPROJP * DIN + 255) / 256, 256, 0, stream>>>(x_proj_w, w_x_h);
    mamba_cvt_f16  <<<(C_DIM * DIN + 255) / 256, 256, 0, stream>>>(out_proj_w, w_out_h, C_DIM * DIN);
    mamba_aneg     <<<(DIN * NST + 255) / 256, 256, 0, stream>>>(A_log, aneg);

    // layernorm -> xs_f16[L,128]
    mamba_layernorm<<<L_SEQ / 256, 256, 0, stream>>>(x, ln_w, ln_b, xs_h);

    // in_proj: [L,128] x [512,128]^T -> xz[L,512]   (NT=4: 64 cols per block-col)
    mamba_gemm_wmma<128, 4, false><<<dim3(L_SEQ / 64, 512 / 64), 128, 0, stream>>>(
        xs_h, w_in_h, xz, L_SEQ, 512, 512);

    // causal depthwise conv + SiLU -> xin (f32 + f16)
    mamba_conv_silu<<<L_SEQ, DIN, 0, stream>>>(xz, conv_w, conv_b, xin_f, xin_h);

    // x_proj: [L,256] x [48,256]^T -> dbl[L,48] (dt|B|C|pad), NT=3 covers all 48 cols
    mamba_gemm_wmma<256, 3, false><<<dim3(L_SEQ / 64, 1), 128, 0, stream>>>(
        xin_h, w_x_h, dbl, L_SEQ, NPROJP, NPROJP);

    // dt_proj + softplus -> dtf[L,256]
    mamba_dtproj<<<L_SEQ, DIN, 0, stream>>>(dbl, dt_proj_w, dt_proj_b, dtf);

    // chunked associative scan (3 passes)
    mamba_scan_carry <<<dim3(DIN, NCHUNK), 32, 0, stream>>>(dtf, xin_f, dbl, aneg, cA, cB);
    mamba_scan_prefix<<<(DIN * NST + 255) / 256, 256, 0, stream>>>(cA, cB, h0);
    mamba_scan_final <<<dim3(DIN, NCHUNK), 32, 0, stream>>>(dtf, xin_f, dbl, aneg, h0,
                                                            xz, D_param, yg);

    // out_proj with transposed store -> d_out [128, 8192] == [1,128,8,32,32]
    mamba_gemm_wmma<256, 4, true><<<dim3(L_SEQ / 64, C_DIM / 64), 128, 0, stream>>>(
        yg, w_out_h, out, L_SEQ, C_DIM, C_DIM);
}
